// Net_APPNP_71889162600827
// MI455X (gfx1250) — matrix operand order repaired and checked by reading, not compiled, not run on hardware
//
#include <hip/hip_runtime.h>
#include <hip/hip_bf16.h>

// ---------------------------------------------------------------------------
// Types for CDNA5 WMMA (wave32, 16x16x32 bf16 -> f32)
// ---------------------------------------------------------------------------
typedef __attribute__((ext_vector_type(16))) __bf16 v16bf;
typedef __attribute__((ext_vector_type(8)))  __bf16 v8bf;
typedef __attribute__((ext_vector_type(8)))  float  v8f;

union V16U {
    v16bf v;
    v8bf  h[2];
};

__device__ __forceinline__ unsigned short f32_to_bf16_rne(float f) {
    unsigned int u = __float_as_uint(f);
    unsigned int round_bias = 0x7FFFu + ((u >> 16) & 1u);
    u += round_bias;
    return (unsigned short)(u >> 16);
}

// One 16B async global->LDS copy (CDNA5 GLOBAL_LOAD_ASYNC_TO_LDS_B128,
// GV mode: LDS[vdst_lane] = MEM[vaddr_lane]); tracked by ASYNCcnt.
__device__ __forceinline__ void async_copy_b128(unsigned lds_byte_off,
                                                const void* gaddr) {
    asm volatile("global_load_async_to_lds_b128 %0, %1, off"
                 :
                 : "v"(lds_byte_off), "v"(gaddr)
                 : "memory");
}

__device__ __forceinline__ void wait_asynccnt0() {
    asm volatile("s_wait_asynccnt 0" ::: "memory");
}
// Allow 3 outstanding (the next tile's 3 copies) while this tile is consumed.
__device__ __forceinline__ void wait_asynccnt3() {
    asm volatile("s_wait_asynccnt 3" ::: "memory");
}

// LDS byte offset of a __shared__ pointer (LDS aperture: addr[31:0] = offset).
__device__ __forceinline__ unsigned lds_offset(const void* p) {
    return (unsigned)(unsigned long long)p;
}

// ---------------------------------------------------------------------------
// Elementwise fp32 -> bf16 convert
// ---------------------------------------------------------------------------
__global__ void cvt_bf16_kernel(const float* __restrict__ in,
                                unsigned short* __restrict__ out, size_t n) {
    size_t i = (size_t)blockIdx.x * blockDim.x + threadIdx.x;
    if (i < n) out[i] = f32_to_bf16_rne(in[i]);
}

// W [K][N] fp32 (row major) -> Wt [N][K] bf16 (row major, i.e. transposed)
__global__ void transpose_cvt_kernel(const float* __restrict__ W,
                                     unsigned short* __restrict__ Wt,
                                     int K, int N) {
    int idx = blockIdx.x * blockDim.x + threadIdx.x; // over N*K, output-coalesced
    if (idx >= K * N) return;
    int n = idx / K;
    int k = idx - n * K;
    Wt[idx] = f32_to_bf16_rne(W[(size_t)k * N + n]);
}

// ---------------------------------------------------------------------------
// WMMA GEMM:  Out[M,N] = A[M,K](bf16) * Wt[N,K]^T(bf16) + bias ; optional ReLU
// Block = 256 threads (8 waves). Tile: 128 rows x 64 cols. K stepped by 32.
// Double-buffered LDS tiles filled by async global->LDS B128 copies (ASYNCcnt):
// tile t+1 streams in while tile t feeds 4 v_wmma_f32_16x16x32_bf16 per wave.
// Padded LDS stride 40 halfs (80B, 16B aligned, conflict-light).
// ---------------------------------------------------------------------------
#define LDA 40
#define LDB 40

template <bool RELU, bool OUT_BF16>
__global__ __launch_bounds__(256) void gemm_wmma_bf16(
    const __bf16* __restrict__ A,      // [M,K] bf16
    const __bf16* __restrict__ Bt,     // [N,K] bf16 (transposed weights)
    const float*  __restrict__ bias,   // [N]
    void* __restrict__ Out,            // [M,N] bf16 or f32
    int M, int N, int K) {
    __shared__ __bf16 As[2][128 * LDA];   // 2 x 10240 B
    __shared__ __bf16 Bs[2][64 * LDB];    // 2 x  5120 B

    const int rowBase = blockIdx.x * 128;
    const int colBase = blockIdx.y * 64;

    const int tid  = threadIdx.x;
    const int w    = tid >> 5;        // wave 0..7
    const int lane = tid & 31;
    const int l    = lane & 15;
    const int g    = lane >> 4;

    // Per-thread staging slots (row/seg invariant across K-steps).
    const int rA0 = (tid) >> 2,         segA0 = (tid) & 3;
    const int rA1 = (tid + 256) >> 2,   segA1 = (tid + 256) & 3;
    int grA0 = rowBase + rA0; if (grA0 > M - 1) grA0 = M - 1;
    int grA1 = rowBase + rA1; if (grA1 > M - 1) grA1 = M - 1;
    const int colB = tid >> 2, segB = tid & 3;

    const __bf16* gA0 = A  + (size_t)grA0 * K + segA0 * 8;
    const __bf16* gA1 = A  + (size_t)grA1 * K + segA1 * 8;
    const __bf16* gB  = Bt + (size_t)(colBase + colB) * K + segB * 8;

    unsigned ldsA0[2], ldsA1[2], ldsB[2];
    #pragma unroll
    for (int b = 0; b < 2; ++b) {
        ldsA0[b] = lds_offset(&As[b][rA0 * LDA + segA0 * 8]);
        ldsA1[b] = lds_offset(&As[b][rA1 * LDA + segA1 * 8]);
        ldsB[b]  = lds_offset(&Bs[b][colB * LDB + segB * 8]);
    }

    const int nT = K >> 5;            // K/32 tiles
    v8f acc[4] = {};

    // Prologue: stage tile 0 into buffer 0 (3 async copies -> ASYNCcnt = 3).
    async_copy_b128(ldsA0[0], gA0);
    async_copy_b128(ldsA1[0], gA1);
    async_copy_b128(ldsB[0],  gB);

    for (int t = 0; t < nT; ++t) {
        const int cur = t & 1;
        if (t + 1 < nT) {
            // Stage tile t+1 into the other buffer, then wait only for tile t
            // (in-order completion: ASYNCcnt <= 3 means tile t has landed).
            const int nxt = cur ^ 1;
            const int kk1 = (t + 1) << 5;
            async_copy_b128(ldsA0[nxt], gA0 + kk1);
            async_copy_b128(ldsA1[nxt], gA1 + kk1);
            async_copy_b128(ldsB[nxt],  gB  + kk1);
            wait_asynccnt3();
        } else {
            wait_asynccnt0();
        }
        __syncthreads();   // all waves' tile-t copies visible

        // A fragment: lane l+16g holds row (16w+l), K {8g..8g+7, 16+8g..16+8g+7}
        V16U a;
        a.h[0] = *(const v8bf*)&As[cur][(16 * w + l) * LDA + 8 * g];
        a.h[1] = *(const v8bf*)&As[cur][(16 * w + l) * LDA + 16 + 8 * g];

        #pragma unroll
        for (int c = 0; c < 4; ++c) {
            // B fragment: lane l+16g holds col (16c+l), K {8g..8g+7, 16+8g..16+8g+7}
            V16U b;
            b.h[0] = *(const v8bf*)&Bs[cur][(16 * c + l) * LDB + 8 * g];
            b.h[1] = *(const v8bf*)&Bs[cur][(16 * c + l) * LDB + 16 + 8 * g];
            acc[c] = __builtin_amdgcn_wmma_f32_16x16x32_bf16(
                false, a.v, false, b.v, (short)0, acc[c], false, false);
        }
        __syncthreads();   // all waves done reading buf[cur] before it refills
    }

    // C/D layout: VGPR v, lane l+16g -> row (v + 8g), col l
    #pragma unroll
    for (int c = 0; c < 4; ++c) {
        int n = colBase + 16 * c + l;
        float bv = bias[n];
        #pragma unroll
        for (int v = 0; v < 8; ++v) {
            int row = rowBase + 16 * w + v + 8 * g;
            if (row < M) {
                float val = acc[c][v] + bv;
                if (RELU) val = fmaxf(val, 0.0f);
                if (OUT_BF16)
                    ((unsigned short*)Out)[(size_t)row * N + n] = f32_to_bf16_rne(val);
                else
                    ((float*)Out)[(size_t)row * N + n] = val;
            }
        }
    }
}

// ---------------------------------------------------------------------------
// Degree / propagation kernels
// ---------------------------------------------------------------------------
__global__ void deg_init_kernel(float* __restrict__ deg, int n) {
    int i = blockIdx.x * blockDim.x + threadIdx.x;
    if (i < n) deg[i] = 1.0f;   // self loop
}

__global__ void deg_count_kernel(const int* __restrict__ dst,
                                 float* __restrict__ deg, int E) {
    int e = blockIdx.x * blockDim.x + threadIdx.x;
    if (e < E) atomicAdd(&deg[dst[e]], 1.0f);
}

__global__ void dinv_kernel(const float* __restrict__ deg,
                            float* __restrict__ dinv, int n) {
    int i = blockIdx.x * blockDim.x + threadIdx.x;
    if (i < n) dinv[i] = rsqrtf(deg[i]);
}

__global__ void zero_kernel(float* __restrict__ p, size_t n) {
    size_t i = (size_t)blockIdx.x * blockDim.x + threadIdx.x;
    if (i < n) p[i] = 0.0f;
}

// One wave32 per edge: coalesced float4 gather of h[src], atomic scatter to agg[dst].
__global__ __launch_bounds__(256) void scatter_kernel(
    const int* __restrict__ src, const int* __restrict__ dst,
    const float* __restrict__ dinv, const float* __restrict__ h,
    float* __restrict__ agg, int E) {
    int gt   = blockIdx.x * 256 + threadIdx.x;
    int e    = gt >> 5;
    int lane = threadIdx.x & 31;
    if (e >= E) return;
    int s = src[e];
    int d = dst[e];
    float w = dinv[s] * dinv[d];
    const float4* hs = (const float4*)h + (size_t)s * 64;   // 256 floats = 64 f4
    float* ad = agg + (size_t)d * 256;
    #pragma unroll
    for (int j = 0; j < 2; ++j) {
        int q = lane + 32 * j;
        float4 v = hs[q];
        int base = q * 4;
        atomicAdd(ad + base + 0, w * v.x);
        atomicAdd(ad + base + 1, w * v.y);
        atomicAdd(ad + base + 2, w * v.z);
        atomicAdd(ad + base + 3, w * v.w);
    }
}

// h <- (1-a) * (agg + dinv_i^2 * h) + a * h0   (self-loop folded in analytically)
__global__ void combine_kernel(float* __restrict__ h,
                               const float* __restrict__ agg,
                               const float* __restrict__ h0,
                               const float* __restrict__ dinv,
                               size_t total) {
    size_t i = (size_t)blockIdx.x * blockDim.x + threadIdx.x;
    if (i >= total) return;
    int node = (int)(i >> 8);   // 256 features per node
    float di = dinv[node];
    float hv = h[i];
    h[i] = 0.8f * (agg[i] + di * di * hv) + 0.2f * h0[i];
}

// ---------------------------------------------------------------------------
// Host orchestration
// ---------------------------------------------------------------------------
extern "C" void kernel_launch(void* const* d_in, const int* in_sizes, int n_in,
                              void* d_out, int out_size, void* d_ws, size_t ws_size,
                              hipStream_t stream) {
    const int IN_DIM = 512, HID = 512, OUT = 256;
    const float* x   = (const float*)d_in[0];
    const int*   ei  = (const int*)d_in[1];
    const float* W1  = (const float*)d_in[2];
    const float* b1  = (const float*)d_in[3];
    const float* W2  = (const float*)d_in[4];
    const float* b2  = (const float*)d_in[5];
    const float* W3  = (const float*)d_in[6];
    const float* b3  = (const float*)d_in[7];

    const int M = in_sizes[0] / IN_DIM;       // 50000 nodes
    const int E = in_sizes[1] / 2;            // 1.6M edges
    const int* src = ei;
    const int* dst = ei + E;

    // ---- workspace carve-up (256B aligned) ----
    char* p = (char*)d_ws;
    auto alloc = [&](size_t bytes) -> void* {
        void* r = (void*)p;
        p += (bytes + 255) & ~(size_t)255;
        return r;
    };
    unsigned short* xbf  = (unsigned short*)alloc((size_t)M * IN_DIM * 2); // also reused for h2
    unsigned short* h1bf = (unsigned short*)alloc((size_t)M * HID * 2);
    unsigned short* W1t  = (unsigned short*)alloc((size_t)IN_DIM * HID * 2);
    unsigned short* W2t  = (unsigned short*)alloc((size_t)HID * HID * 2);
    unsigned short* W3t  = (unsigned short*)alloc((size_t)HID * OUT * 2);
    float* h0   = (float*)alloc((size_t)M * OUT * 4);
    float* agg  = (float*)alloc((size_t)M * OUT * 4);
    float* deg  = (float*)alloc((size_t)M * 4);
    float* dinv = (float*)alloc((size_t)M * 4);
    float* h    = (float*)d_out;              // [M, 256] fp32 — evolves in place

    // ---- 1. bf16 conversions ----
    {
        size_t n = (size_t)M * IN_DIM;
        cvt_bf16_kernel<<<(unsigned)((n + 255) / 256), 256, 0, stream>>>(x, xbf, n);
    }
    transpose_cvt_kernel<<<(IN_DIM * HID + 255) / 256, 256, 0, stream>>>(W1, W1t, IN_DIM, HID);
    transpose_cvt_kernel<<<(HID * HID + 255) / 256, 256, 0, stream>>>(W2, W2t, HID, HID);
    transpose_cvt_kernel<<<(HID * OUT + 255) / 256, 256, 0, stream>>>(W3, W3t, HID, OUT);

    // ---- 2. MLP encoder via WMMA ----
    dim3 blk(256);
    int mBlocks = (M + 127) / 128;
    // h1 = relu(x@W1+b1), bf16 out
    gemm_wmma_bf16<true, true><<<dim3(mBlocks, HID / 64), blk, 0, stream>>>(
        (const __bf16*)xbf, (const __bf16*)W1t, b1, (void*)h1bf, M, HID, IN_DIM);
    // h2 = relu(h1@W2+b2), bf16 out (reuse xbf buffer)
    gemm_wmma_bf16<true, true><<<dim3(mBlocks, HID / 64), blk, 0, stream>>>(
        (const __bf16*)h1bf, (const __bf16*)W2t, b2, (void*)xbf, M, HID, HID);
    // h0 = h2@W3+b3, fp32 out
    gemm_wmma_bf16<false, false><<<dim3(mBlocks, OUT / 64), blk, 0, stream>>>(
        (const __bf16*)xbf, (const __bf16*)W3t, b3, (void*)h0, M, OUT, HID);

    // h := h0 (propagation state lives in d_out)
    hipMemcpyAsync(h, h0, (size_t)M * OUT * sizeof(float),
                   hipMemcpyDeviceToDevice, stream);

    // ---- 3. GCN normalization ----
    deg_init_kernel<<<(M + 255) / 256, 256, 0, stream>>>(deg, M);
    deg_count_kernel<<<(E + 255) / 256, 256, 0, stream>>>(dst, deg, E);
    dinv_kernel<<<(M + 255) / 256, 256, 0, stream>>>(deg, dinv, M);

    // ---- 4. APPNP propagation, K = 10 ----
    size_t total = (size_t)M * OUT;
    unsigned zeroBlocks    = (unsigned)((total + 255) / 256);
    unsigned scatterBlocks = (unsigned)(((size_t)E * 32 + 255) / 256);
    for (int k = 0; k < 10; ++k) {
        zero_kernel<<<zeroBlocks, 256, 0, stream>>>(agg, total);
        scatter_kernel<<<scatterBlocks, 256, 0, stream>>>(src, dst, dinv, h, agg, E);
        combine_kernel<<<zeroBlocks, 256, 0, stream>>>(h, agg, h0, dinv, total);
    }
}
